// Attention_46462956208325
// MI455X (gfx1250) — matrix operand order repaired and checked by reading, not compiled, not run on hardware
//
#include <hip/hip_runtime.h>
#include <hip/hip_bf16.h>
#include <stdint.h>

#define B_     4
#define DEC_   256
#define ENC_   1024
#define HID_   512
#define UNITS_ 128

typedef __attribute__((ext_vector_type(16))) _Float16 v16h;
typedef __attribute__((ext_vector_type(8)))  _Float16 v8h;
typedef __attribute__((ext_vector_type(4)))  _Float16 v4h;
typedef __attribute__((ext_vector_type(8)))  float    v8f;

__device__ __forceinline__ float fast_tanh(float x) {
#if __has_builtin(__builtin_amdgcn_tanhf)
  return __builtin_amdgcn_tanhf(x);                 // v_tanh_f32 (gfx1250 TRANS op)
#else
  float e = __builtin_amdgcn_exp2f(x * 2.8853900817779268f);  // e^(2x)
  return 1.0f - 2.0f / (e + 1.0f);
#endif
}

// ---------------------------------------------------------------------------
// Kernel 0: one-time f32 -> f16 conversion into workspace (packed 4-wide).
// ---------------------------------------------------------------------------
__global__ void __launch_bounds__(256)
cvt_f32_to_f16(const float* __restrict__ src, _Float16* __restrict__ dst, int n4) {
  const int stride = gridDim.x * blockDim.x;
  for (int i = blockIdx.x * blockDim.x + threadIdx.x; i < n4; i += stride) {
    float4 v = ((const float4*)src)[i];
    v4h h;
    h[0] = (_Float16)v.x; h[1] = (_Float16)v.y;
    h[2] = (_Float16)v.z; h[3] = (_Float16)v.w;
    ((v4h*)dst)[i] = h;
  }
}

// ---------------------------------------------------------------------------
// Kernel 1: Y(rows,128) = X(rows,512) @ W(512,128), f16 in / f32 accumulate.
// One wave per 16x16 tile. A (16-bit 16x32): lane l -> m=l&15, h=l>>4,
// elems 0..7 = K 8h+j, elems 8..15 = K 16+8h+j.  B (32x16): lane -> N=m,
// elems = same K order as A.  C/D: vgpr r -> M=r+8h, N=m.
// ---------------------------------------------------------------------------
__global__ void __launch_bounds__(32)
proj_f16_wmma(const _Float16* __restrict__ X, const _Float16* __restrict__ W,
              float* __restrict__ Y) {
  const int lane = threadIdx.x;
  const int m = lane & 15, hh = lane >> 4;
  const int row0 = blockIdx.x * 16;
  const int u0   = blockIdx.y * 16;

  v8f acc = {};
  const _Float16* xrow = X + (size_t)(row0 + m) * HID_;
  for (int kb = 0; kb < HID_; kb += 32) {
    v8h lo = *(const v8h*)(xrow + kb + 8 * hh);        // 16 B load
    v8h hi = *(const v8h*)(xrow + kb + 16 + 8 * hh);   // 16 B load
    v16h a = __builtin_shufflevector(lo, hi, 0, 1, 2, 3, 4, 5, 6, 7,
                                     8, 9, 10, 11, 12, 13, 14, 15);
    v16h bm;
    for (int s = 0; s < 16; ++s)
      bm[s] = W[(size_t)(kb + (s < 8 ? 8 * hh + s : 16 + 8 * hh + (s - 8))) * UNITS_ + u0 + m];
    acc = __builtin_amdgcn_wmma_f32_16x16x32_f16(false, a, false, bm,
                                                 (short)0, acc, false, false);
  }
#pragma unroll
  for (int r = 0; r < 8; ++r)
    Y[(size_t)(row0 + r + 8 * hh) * UNITS_ + u0 + m] = acc[r];
}

// ---------------------------------------------------------------------------
// Kernel 2: Bahdanau scores + mask + softmax.
// Block = (batch, 16 query rows); 256 threads = 16 lanes x 16 rows.
// k-tiles staged via GLOBAL_LOAD_ASYNC_TO_LDS_B128 (ASYNCcnt), reused by all
// 16 query rows. Scores live in 64 VGPRs/thread; softmax via width-16
// shuffles. Emits both f32 attn (output) and f16 attn (for context GEMM).
// ---------------------------------------------------------------------------
__global__ void __launch_bounds__(256)
score_softmax(const float* __restrict__ qproj, const float* __restrict__ kproj,
              const uint8_t* __restrict__ mask, const float* __restrict__ scale,
              float* __restrict__ attn, _Float16* __restrict__ attn16) {
  __shared__ float s_q[16][UNITS_];          // 8 KB
  __shared__ float s_scale[UNITS_];          // 0.5 KB
  __shared__ float s_k[64][UNITS_ + 4];      // 33 KB, padded vs bank conflicts

  const int t  = threadIdx.x;
  const int r  = t >> 4;                     // query row within tile
  const int c  = t & 15;                     // enc sub-lane
  const int b  = blockIdx.x / (DEC_ / 16);
  const int q0 = (blockIdx.x % (DEC_ / 16)) * 16;

  for (int i = t; i < 16 * UNITS_; i += 256)
    s_q[i / UNITS_][i % UNITS_] =
        qproj[(size_t)(b * DEC_ + q0 + i / UNITS_) * UNITS_ + (i % UNITS_)];
  for (int i = t; i < UNITS_; i += 256) s_scale[i] = scale[i];

  float sc[64];
  for (int ch = 0; ch < ENC_ / 64; ++ch) {
    __syncthreads();  // previous chunk fully consumed
    // Async memory->LDS staging: 64 rows x 32 x 16B = 32 KB, no VGPR bounce.
    for (int i = t; i < 64 * 32; i += 256) {
      const int row = i >> 5, seg = i & 31;
      uint32_t lds = (uint32_t)(uintptr_t)&s_k[row][seg * 4];
      uint64_t ga  = (uint64_t)(uintptr_t)(kproj +
          (size_t)(b * ENC_ + ch * 64 + row) * UNITS_ + seg * 4);
      asm volatile("global_load_async_to_lds_b128 %0, %1, off"
                   :: "v"(lds), "v"(ga) : "memory");
    }
    asm volatile("s_wait_asynccnt 0" ::: "memory");
    __syncthreads();  // all waves' async copies visible

#pragma unroll
    for (int j = 0; j < 4; ++j) {
      const int eloc = c + 16 * j;
      float s = 0.f;
#pragma unroll 8
      for (int u = 0; u < UNITS_; ++u)
        s += s_scale[u] * fast_tanh(s_q[r][u] + s_k[eloc][u]);
      const int e = ch * 64 + eloc;
      sc[ch * 4 + j] = mask[b * ENC_ + e] ? s : -1e9f;
    }
  }

  // Softmax across 16-lane groups x 64 registers (wave32-safe width-16).
  float mx = -3.4e38f;
#pragma unroll
  for (int i = 0; i < 64; ++i) mx = fmaxf(mx, sc[i]);
  for (int off = 8; off >= 1; off >>= 1) mx = fmaxf(mx, __shfl_xor(mx, off, 16));

  float sum = 0.f;
#pragma unroll
  for (int i = 0; i < 64; ++i) {
    float p = __builtin_amdgcn_exp2f((sc[i] - mx) * 1.4426950408889634f);
    sc[i] = p;
    sum += p;
  }
  for (int off = 8; off >= 1; off >>= 1) sum += __shfl_xor(sum, off, 16);
  const float inv = 1.0f / sum;

  float*    arow   = attn   + (size_t)(b * DEC_ + q0 + r) * ENC_;
  _Float16* arow16 = attn16 + (size_t)(b * DEC_ + q0 + r) * ENC_;
#pragma unroll
  for (int i = 0; i < 64; ++i) {
    const int e = (i >> 2) * 64 + c + 16 * (i & 3);
    const float w = sc[i] * inv;
    arow[e]   = w;
    arow16[e] = (_Float16)w;
  }
}

// ---------------------------------------------------------------------------
// Kernel 3: context(b,256,512) = attn16(b,256,1024) @ value16(b,1024,512).
// One wave per 16x16 tile, K=1024 in 32 WMMA steps, pure f16 loads.
// ---------------------------------------------------------------------------
__global__ void __launch_bounds__(32)
context_wmma(const _Float16* __restrict__ attn16,
             const _Float16* __restrict__ value16, float* __restrict__ ctx) {
  const int lane = threadIdx.x;
  const int m = lane & 15, hh = lane >> 4;
  const int q0 = blockIdx.x * 16;
  const int h0 = blockIdx.y * 16;
  const int b  = blockIdx.z;

  v8f acc = {};
  const _Float16* arow = attn16 + (size_t)(b * DEC_ + q0 + m) * ENC_;
  for (int kb = 0; kb < ENC_; kb += 32) {
    v8h lo = *(const v8h*)(arow + kb + 8 * hh);
    v8h hi = *(const v8h*)(arow + kb + 16 + 8 * hh);
    v16h a = __builtin_shufflevector(lo, hi, 0, 1, 2, 3, 4, 5, 6, 7,
                                     8, 9, 10, 11, 12, 13, 14, 15);
    v16h bm;
    for (int s = 0; s < 16; ++s)
      bm[s] = value16[(size_t)(b * ENC_ + kb + (s < 8 ? 8 * hh + s : 16 + 8 * hh + (s - 8))) * HID_ + h0 + m];
    acc = __builtin_amdgcn_wmma_f32_16x16x32_f16(false, a, false, bm,
                                                 (short)0, acc, false, false);
  }
#pragma unroll
  for (int r = 0; r < 8; ++r)
    ctx[(size_t)(b * DEC_ + q0 + r + 8 * hh) * HID_ + h0 + m] = acc[r];
}

// ---------------------------------------------------------------------------
extern "C" void kernel_launch(void* const* d_in, const int* in_sizes, int n_in,
                              void* d_out, int out_size, void* d_ws, size_t ws_size,
                              hipStream_t stream) {
  (void)in_sizes; (void)n_in; (void)out_size; (void)ws_size;
  const float*   query = (const float*)d_in[0];   // (B,DEC,HID)
  const float*   value = (const float*)d_in[1];   // (B,ENC,HID)
  const uint8_t* mask  = (const uint8_t*)d_in[2]; // (B,ENC) bool
  const float*   W1    = (const float*)d_in[3];   // (HID,U)
  const float*   W2    = (const float*)d_in[4];   // (HID,U)
  const float*   scale = (const float*)d_in[5];   // (U,)

  float* ctx_out  = (float*)d_out;                         // B*DEC*HID
  float* attn_out = ctx_out + (size_t)B_ * DEC_ * HID_;    // B*DEC*ENC

  // Workspace layout (all chunks 16B-aligned):
  char* ws = (char*)d_ws;
  const size_t NQ = (size_t)B_ * DEC_ * HID_;   // 524288
  const size_t NV = (size_t)B_ * ENC_ * HID_;   // 2097152
  const size_t NW = (size_t)HID_ * UNITS_;      // 65536
  const size_t NA = (size_t)B_ * DEC_ * ENC_;   // 1048576
  _Float16* qf16   = (_Float16*)ws;              ws += NQ * 2;
  _Float16* vf16   = (_Float16*)ws;              ws += NV * 2;
  _Float16* w1f16  = (_Float16*)ws;              ws += NW * 2;
  _Float16* w2f16  = (_Float16*)ws;              ws += NW * 2;
  _Float16* attn16 = (_Float16*)ws;              ws += NA * 2;
  float*    qproj  = (float*)ws;                 ws += (size_t)B_ * DEC_ * UNITS_ * 4;
  float*    kproj  = (float*)ws;

  // One-time f16 mirrors of the GEMM operands.
  cvt_f32_to_f16<<<dim3((int)(NQ / 4 / 256)), 256, 0, stream>>>(query, qf16, (int)(NQ / 4));
  cvt_f32_to_f16<<<dim3((int)(NV / 4 / 256)), 256, 0, stream>>>(value, vf16, (int)(NV / 4));
  cvt_f32_to_f16<<<dim3((int)(NW / 4 / 64)),  64,  0, stream>>>(W1, w1f16, (int)(NW / 4));
  cvt_f32_to_f16<<<dim3((int)(NW / 4 / 64)),  64,  0, stream>>>(W2, w2f16, (int)(NW / 4));

  proj_f16_wmma<<<dim3(B_ * DEC_ / 16, UNITS_ / 16), 32, 0, stream>>>(qf16, w1f16, qproj);
  proj_f16_wmma<<<dim3(B_ * ENC_ / 16, UNITS_ / 16), 32, 0, stream>>>(vf16, w2f16, kproj);
  score_softmax<<<dim3(B_ * DEC_ / 16), 256, 0, stream>>>(qproj, kproj, mask, scale,
                                                          attn_out, attn16);
  context_wmma<<<dim3(DEC_ / 16, HID_ / 16, B_), 32, 0, stream>>>(attn16, vf16, ctx_out);
}
